// CConvLayer_50216757625045
// MI455X (gfx1250) — matrix operand order repaired and checked by reading, not compiled, not run on hardware
//
#include <hip/hip_runtime.h>

typedef __attribute__((ext_vector_type(16))) _Float16 v16h;
typedef __attribute__((ext_vector_type(8)))  float    v8f;
typedef __attribute__((ext_vector_type(8)))  unsigned v8u;

#define P_NB   32     // neighbors per point (= WMMA K for phase A)
#define INC    64     // input channels
#define OUTC   64     // output channels
#define KH     4
#define KW     4
#define GB     16     // kernel bins = KH*KW
#define BN     16     // points per workgroup (= WMMA M for phase B)
#define TPB    128    // 4 waves of 32
#define NWAVE  4

#define KFRAG_HALVES (32 * 4 * 32 * 16)       // chunks * tiles * lanes * halves
#define KFRAG_BYTES  (KFRAG_HALVES * 2)       // 131072 B
#define FEAT_OFF     KFRAG_BYTES              // f16 features follow kernel frags

__device__ __forceinline__ int imin(int a, int b) { return a < b ? a : b; }
__device__ __forceinline__ int imax(int a, int b) { return a > b ? a : b; }

// 16-bit A-matrix (16xK) k index for dword j (pair k,k+1), same order as B:
// lanes 0-15: K = 0..15; lanes 16-31: K = 16..31
__device__ __forceinline__ int kmapA(int laneHi, int j) {
    return laneHi ? 16 + 2 * j : 2 * j;
}
// 16-bit B-matrix (32x16) k index for dword j:
// lanes 0-15 hold K=0..15, lanes 16-31 hold K=16..31 (2 per VGPR)
__device__ __forceinline__ int kmapB(int laneHi, int j) {
    return laneHi ? 16 + 2 * j : 2 * j;
}

union H2U { _Float16 h[2]; unsigned u; };

// ---- Prep 1: fp32 kernel [g][i][o] -> f16 WMMA-B fragments in ws ----------
// Layout: wsK dword D = ((chunk*4 + tile)*32 + lane)*8 + j  holds halves
//         (k, k+1) with k = chunk*32 + kmapB(lane>>4, j), o = tile*16+lane%16
__global__ __launch_bounds__(256) void prep_kern_frags(
    const float* __restrict__ kern, unsigned* __restrict__ wsK)
{
    int D = blockIdx.x * 256 + threadIdx.x;          // 32768 dwords total
    if (D >= KFRAG_HALVES / 2) return;
    int j     = D & 7;
    int lane  = (D >> 3) & 31;
    int tile  = (D >> 8) & 3;
    int chunk = D >> 10;
    int k = chunk * 32 + kmapB((lane >> 4) & 1, j);
    int o = tile * 16 + (lane & 15);
    H2U v;
    v.h[0] = (_Float16)kern[(k)     * OUTC + o];
    v.h[1] = (_Float16)kern[(k + 1) * OUTC + o];
    wsK[D] = v.u;
}

// ---- Prep 2: fp32 features -> f16 copy in ws (halves the gather bytes) ----
__global__ __launch_bounds__(256) void prep_feat_f16(
    const float* __restrict__ f, unsigned* __restrict__ fh, int npairs)
{
    int i = blockIdx.x * 256 + threadIdx.x;
    if (i >= npairs) return;
    float2 x = reinterpret_cast<const float2*>(f)[i];
    H2U v;
    v.h[0] = (_Float16)x.x;
    v.h[1] = (_Float16)x.y;
    fh[i] = v.u;
}

// ---------------------------- Main kernel ----------------------------------
template <bool KWS, bool FWS>
__global__ __launch_bounds__(TPB) void cconv_wmma_kernel(
    const float*    __restrict__ features,   // [N,64] fp32
    const int*      __restrict__ receivers,  // [N,32]
    const float*    __restrict__ relpos,     // [N,32,2]
    const float*    __restrict__ wsup_p,     // scalar
    const int*      __restrict__ a_p,        // scalar (window exponent)
    const float*    __restrict__ kern,       // [16,64,64] = [g][i][o] fp32
    const float*    __restrict__ bias,       // [64]
    const _Float16* __restrict__ kernH,      // ws: swizzled f16 B-fragments
    const _Float16* __restrict__ featH,      // ws: f16 features
    float*          __restrict__ out,        // [N,64]
    int N)
{
    __shared__ float    s_w4 [BN * P_NB * 4];             // 8 KB
    __shared__ unsigned s_bin[BN * P_NB];                 // 2 KB
    __shared__ int      s_rcv[BN * P_NB];                 // 2 KB
    __shared__ __align__(16) _Float16 s_M[BN * GB * INC]; // 32 KB  [pt][g][i]

    const int tid    = threadIdx.x;
    const int wave   = tid >> 5;
    const int lane   = tid & 31;
    const int laneHi = lane >> 4;
    const int lane16 = lane & 15;
    const int nbase  = blockIdx.x * BN;

    const float inv_ws = 1.0f / wsup_p[0];
    const int   aexp   = a_p[0];

    // ---------------- Phase 0: per-(point, neighbor) geometry ----------------
    for (int idx = tid; idx < BN * P_NB; idx += TPB) {
        int pt = idx >> 5, p = idx & 31;
        int n  = imin(nbase + pt, N - 1);
        s_rcv[idx] = receivers[n * P_NB + p];
        float2 rp = reinterpret_cast<const float2*>(relpos)[n * P_NB + p];
        float r0 = rp.x * inv_ws;
        float r1 = rp.y * inv_ws;
        float d2   = r0 * r0 + r1 * r1;
        float base = fmaxf(1.0f - d2, 0.0f);
        float win  = 1.0f;
        for (int e = 0; e < aexp; ++e) win *= base;
        float gy = fminf(fmaxf((r0 + 1.0f) * 0.5f * (KH - 1), 0.0f), (float)(KH - 1));
        float gx = fminf(fmaxf((r1 + 1.0f) * 0.5f * (KW - 1), 0.0f), (float)(KW - 1));
        int   y0 = imin(imax((int)floorf(gy), 0), KH - 2);
        int   x0 = imin(imax((int)floorf(gx), 0), KW - 2);
        float fy = gy - (float)y0;
        float fx = gx - (float)x0;
        s_w4[idx * 4 + 0] = (1.0f - fy) * (1.0f - fx) * win;
        s_w4[idx * 4 + 1] = (1.0f - fy) * fx          * win;
        s_w4[idx * 4 + 2] = fy          * (1.0f - fx) * win;
        s_w4[idx * 4 + 3] = fy          * fx          * win;
        unsigned g00 = (unsigned)(y0 * KW + x0);
        s_bin[idx] = g00 | ((g00 + 1) << 8) | ((g00 + KW) << 16) | ((g00 + KW + 1) << 24);
    }
    __syncthreads();

    // --------- Phase A: per point, M[16g x 64i] = W[16x32] * F[32x64] --------
    for (int pi = 0; pi < BN / NWAVE; ++pi) {
        const int pt = wave * (BN / NWAVE) + pi;

        // A fragment: row g = lane16, columns = neighbors (A k-map)
        v16h afrag;
        const unsigned g = (unsigned)lane16;
#pragma unroll
        for (int j = 0; j < 8; ++j) {
            const int k0 = kmapA(laneHi, j);
#pragma unroll
            for (int hh = 0; hh < 2; ++hh) {
                const int p = k0 + hh;
                const unsigned b = s_bin[pt * P_NB + p];
                const float*   w = &s_w4[(pt * P_NB + p) * 4];
                float acc = 0.0f;
                acc += (( b        & 0xFFu) == g) ? w[0] : 0.0f;
                acc += (((b >>  8) & 0xFFu) == g) ? w[1] : 0.0f;
                acc += (((b >> 16) & 0xFFu) == g) ? w[2] : 0.0f;
                acc += (((b >> 24) & 0xFFu) == g) ? w[3] : 0.0f;
                afrag[2 * j + hh] = (_Float16)acc;
            }
        }

#pragma unroll
        for (int t = 0; t < 4; ++t) {               // 16-channel tiles
            const int ch = t * 16 + lane16;
            v16h bfrag;                              // col n = ch, rows = neighbors
#pragma unroll
            for (int j = 0; j < 8; ++j) {
                const int k0 = kmapB(laneHi, j);
#pragma unroll
                for (int hh = 0; hh < 2; ++hh) {
                    const int rv = s_rcv[pt * P_NB + k0 + hh];
                    // half-wave shares neighbor -> contiguous channels (coalesced)
                    if (FWS) bfrag[2 * j + hh] = featH[rv * INC + ch];
                    else     bfrag[2 * j + hh] = (_Float16)features[rv * INC + ch];
                }
            }
            v8f c = {};
            c = __builtin_amdgcn_wmma_f32_16x16x32_f16(
                    false, afrag, false, bfrag, (short)0, c, false, false);
            // C rows = bins: m = r + 8*laneHi, col i = t*16 + lane16
#pragma unroll
            for (int r = 0; r < 8; ++r) {
                const int gg = r + 8 * laneHi;
                s_M[(pt * GB + gg) * INC + t * 16 + lane16] = (_Float16)c[r];
            }
        }
    }
    __syncthreads();

    // ------ Phase B: out[16pt x 64o] = Mflat[16 x 1024] * Kflat[1024 x 64] ---
    {
        v8f c = {};
        const int ocol = wave * 16 + lane16;
        const int m    = lane16;                      // point row (both halves)
        for (int chunk = 0; chunk < 32; ++chunk) {
            union { v16h h; v8u u; uint4 q[2]; } af, bf;
#pragma unroll
            for (int j = 0; j < 8; ++j) {
                const int kkA = chunk * 32 + kmapA(laneHi, j);
                af.u[j] = *reinterpret_cast<const unsigned*>(
                              &s_M[m * (GB * INC) + kkA]);
            }
            if (KWS) {
                // pre-swizzled f16 fragments: two b128 loads per lane
                const uint4* kf = reinterpret_cast<const uint4*>(kernH)
                                  + (size_t)((chunk * 4 + wave) * 32 + lane) * 2;
                bf.q[0] = kf[0];
                bf.q[1] = kf[1];
            } else {
#pragma unroll
                for (int j = 0; j < 8; ++j) {
                    const int kkB = chunk * 32 + kmapB(laneHi, j);
                    bf.h[2 * j + 0] = (_Float16)kern[(kkB)     * OUTC + ocol];
                    bf.h[2 * j + 1] = (_Float16)kern[(kkB + 1) * OUTC + ocol];
                }
            }
            c = __builtin_amdgcn_wmma_f32_16x16x32_f16(
                    false, af.h, false, bf.h, (short)0, c, false, false);
        }
        const float bv = bias[ocol];
#pragma unroll
        for (int r = 0; r < 8; ++r) {
            const int mm = r + 8 * laneHi;
            const int n  = nbase + mm;
            if (n < N)
                out[n * OUTC + ocol] = c[r] * (1.0f / (float)P_NB) + bv;
        }
    }
}

extern "C" void kernel_launch(void* const* d_in, const int* in_sizes, int n_in,
                              void* d_out, int out_size, void* d_ws, size_t ws_size,
                              hipStream_t stream) {
    const float* features  = (const float*)d_in[0];
    const int*   receivers = (const int*)  d_in[1];
    const float* relpos    = (const float*)d_in[2];
    const float* wsup      = (const float*)d_in[3];
    const int*   a         = (const int*)  d_in[4];
    const float* kern      = (const float*)d_in[5];
    const float* bias      = (const float*)d_in[6];
    float*       out       = (float*)d_out;

    const int N      = in_sizes[0] / INC;              // 50000
    const int blocks = (N + BN - 1) / BN;              // 3125

    const size_t needK = (size_t)KFRAG_BYTES;
    const size_t needF = needK + (size_t)N * INC * sizeof(_Float16);
    const bool   kws   = ws_size >= needK;
    const bool   fws   = ws_size >= needF;

    unsigned*  wsK   = (unsigned*)d_ws;
    _Float16*  featH = (_Float16*)((char*)d_ws + FEAT_OFF);

    if (kws) {
        hipLaunchKernelGGL(prep_kern_frags, dim3((KFRAG_HALVES / 2 + 255) / 256),
                           dim3(256), 0, stream, kern, wsK);
    }
    if (fws) {
        const int npairs = N * INC / 2;
        hipLaunchKernelGGL(prep_feat_f16, dim3((npairs + 255) / 256),
                           dim3(256), 0, stream, features, (unsigned*)featH,
                           npairs);
    }

    if (fws) {
        hipLaunchKernelGGL((cconv_wmma_kernel<true, true>), dim3(blocks),
                           dim3(TPB), 0, stream, features, receivers, relpos,
                           wsup, a, kern, bias, (const _Float16*)wsK, featH,
                           out, N);
    } else if (kws) {
        hipLaunchKernelGGL((cconv_wmma_kernel<true, false>), dim3(blocks),
                           dim3(TPB), 0, stream, features, receivers, relpos,
                           wsup, a, kern, bias, (const _Float16*)wsK, featH,
                           out, N);
    } else {
        hipLaunchKernelGGL((cconv_wmma_kernel<false, false>), dim3(blocks),
                           dim3(TPB), 0, stream, features, receivers, relpos,
                           wsup, a, kern, bias, (const _Float16*)wsK, featH,
                           out, N);
    }
}
